// LSTMNetwork_10694468567567
// MI455X (gfx1250) — matrix operand order repaired and checked by reading, not compiled, not run on hardware
//
#include <hip/hip_runtime.h>

// ---------------------------------------------------------------------------
// 3-layer LSTM (H=1024, T=4096, IN=64, OUT=1), batch=1, fp32 reference.
//   * Input projections  pre = X @ W_ih^T + (b_ih+b_hh)  -> bf16 WMMA GEMM,
//       each wave computes a 16x64 strip; statically-named double-buffered
//       fragments (all register-resident) so loads overlap WMMAs.
//   * Recurrence: persistent kernel, W_hh sharded across 256 blocks into LDS
//       (bf16, packed), h double-buffered in global and staged into LDS each
//       step via GLOBAL_LOAD_ASYNC_TO_LDS_B128.
//   * Final linear       out = hs2[T-1] @ w_lin^T + b_lin
// ---------------------------------------------------------------------------

typedef unsigned short ushort_t;
typedef __attribute__((ext_vector_type(16))) __bf16 v16bf;
typedef __attribute__((ext_vector_type(8)))  float  v8f;
typedef int v4i_t __attribute__((vector_size(16)));

#define AS_GLOBAL __attribute__((address_space(1)))
#define AS_SHARED __attribute__((address_space(3)))

#define T_STEPS 4096
#define H_DIM   1024
#define G_DIM   4096   // 4*H
#define IN_DIM  64

#define REC_BLOCKS 256
#define REC_UNITS  4    // hidden units per block (256*4 = 1024)

#if defined(__HIP_DEVICE_COMPILE__) && defined(__has_builtin)
#if __has_builtin(__builtin_amdgcn_global_load_async_to_lds_b128)
#define HAVE_ASYNC_COPY 1
#endif
#if __has_builtin(__builtin_amdgcn_s_wait_asynccnt)
#define HAVE_WAIT_ASYNC 1
#endif
#endif

// ---------------- helpers ----------------
__device__ __forceinline__ ushort_t f32_to_bf16(float x) {
  unsigned b = __builtin_bit_cast(unsigned, x);
  return (ushort_t)((b + 0x7FFFu + ((b >> 16) & 1u)) >> 16);
}
__device__ __forceinline__ unsigned pack_bf16x2(float lo, float hi) {
  return (unsigned)f32_to_bf16(lo) | ((unsigned)f32_to_bf16(hi) << 16);
}
__device__ __forceinline__ float bf16_lo(unsigned u) {
  return __builtin_bit_cast(float, u << 16);
}
__device__ __forceinline__ float bf16_hi(unsigned u) {
  return __builtin_bit_cast(float, u & 0xFFFF0000u);
}
__device__ __forceinline__ float sigmoidf_(float x) {
  return 1.0f / (1.0f + __expf(-x));
}

// ---------------- init: zero barriers + h ping-pong buffers ----------------
__global__ void init_ws_kernel(unsigned* bars, float* hbufs /*3*2*1024*/) {
  int i = blockIdx.x * blockDim.x + threadIdx.x;
  if (i < 3 * 2 * H_DIM) hbufs[i] = 0.0f;
  if (i < 3) bars[i * 16] = 0u;   // 64B-strided barrier slots
}

// ---------------- fp32 -> bf16 cast ----------------
__global__ void cast_f32_bf16_kernel(const float* __restrict__ src,
                                     ushort_t* __restrict__ dst, int n) {
  int i = blockIdx.x * blockDim.x + threadIdx.x;
  if (i < n) dst[i] = f32_to_bf16(src[i]);
}

// ---------------- WMMA GEMM: C[M,N] = A[M,K] * B[N,K]^T + bias[N] ----------
// A, B bf16 row-major; C fp32 row-major. Each wave: one 16x64 strip
// (4 sub-tiles sharing the A fragment -> 4 v_wmma per 32-k step).
// Loop unrolled by 64-k with two statically-named fragment sets so next
// half-step loads are issued before current half-step WMMAs (no arrays with
// runtime indices -> everything register-resident).
// Lane layouts used here (contiguous k, same for A and B):
//   A 16x32 bf16 : lane<16 -> M=lane, K k0+0..15;
//                  lane>=16 -> M=lane-16, K k0+16..31
//   B 32x16 bf16 : lane<16 -> N=lane, K k0+0..15; lane>=16 -> N=lane-16, K k0+16..31
//   C 16x16 f32  : VGPR r: lane<16 -> (M=r, N=lane); lane>=16 -> (M=8+r, N=lane-16)
union frag_t { uint4 q[2]; v16bf v; };

__device__ __forceinline__ void load_afrag(frag_t& f, const ushort_t* p) {
  f.q[0] = *(const uint4*)(p);
  f.q[1] = *(const uint4*)(p + 8);
}
__device__ __forceinline__ void load_bfrag(frag_t& f, const ushort_t* p) {
  f.q[0] = *(const uint4*)(p);
  f.q[1] = *(const uint4*)(p + 8);
}

__global__ __launch_bounds__(256)
void gemm_bf16_wmma_kernel(const ushort_t* __restrict__ A,
                           const ushort_t* __restrict__ B,
                           const float* __restrict__ b_ih,
                           const float* __restrict__ b_hh,
                           float* __restrict__ C,
                           int M, int N, int K) {
  const int wid    = (blockIdx.x * blockDim.x + threadIdx.x) >> 5;
  const int lane   = threadIdx.x & 31;
  const int stripsN = N >> 6;                 // 64-wide N strips
  const int tm = wid / stripsN;
  const int tn = wid % stripsN;
  if (tm * 16 >= M) return;

  const int lrow = lane & 15;
  const int lhi  = lane >> 4;

  v8f c[4];
#pragma unroll
  for (int s = 0; s < 4; ++s) {
    const int n = tn * 64 + s * 16 + lrow;
    const float bias = b_ih[n] + b_hh[n];
#pragma unroll
    for (int r = 0; r < 8; ++r) c[s][r] = bias;
  }

  const ushort_t* arow  = A + (size_t)(tm * 16 + lrow) * K + lhi * 16;
  const ushort_t* brow0 = B + (size_t)(tn * 64 + lrow) * K + lhi * 16;
  const size_t bstep = (size_t)16 * K;        // next 16-N sub-tile

  frag_t a0, a1, b0[4], b1[4];

  // prologue: fragments for k = 0
  load_afrag(a0, arow);
#pragma unroll
  for (int s = 0; s < 4; ++s) load_bfrag(b0[s], brow0 + (size_t)s * bstep);

  // K is 64 or 1024 (multiple of 64): two 32-k halves per iteration
  for (int k0 = 0; k0 < K; k0 += 64) {
    // issue loads for second half before consuming first-half fragments
    load_afrag(a1, arow + k0 + 32);
#pragma unroll
    for (int s = 0; s < 4; ++s)
      load_bfrag(b1[s], brow0 + (size_t)s * bstep + k0 + 32);

#pragma unroll
    for (int s = 0; s < 4; ++s)
      c[s] = __builtin_amdgcn_wmma_f32_16x16x32_bf16(
          false, a0.v, false, b0[s].v, (short)0, c[s], false, false);

    if (k0 + 64 < K) {
      // issue loads for the next iteration's first half
      load_afrag(a0, arow + k0 + 64);
#pragma unroll
      for (int s = 0; s < 4; ++s)
        load_bfrag(b0[s], brow0 + (size_t)s * bstep + k0 + 64);
    }

#pragma unroll
    for (int s = 0; s < 4; ++s)
      c[s] = __builtin_amdgcn_wmma_f32_16x16x32_bf16(
          false, a1.v, false, b1[s].v, (short)0, c[s], false, false);
  }

#pragma unroll
  for (int s = 0; s < 4; ++s) {
    float* crow = C + (size_t)(tm * 16 + lhi * 8) * N + tn * 64 + s * 16 + lrow;
#pragma unroll
    for (int r = 0; r < 8; ++r) crow[(size_t)r * N] = c[s][r];
  }
}

// ---------------- persistent recurrent kernel ----------------
// Block b owns hidden units j0..j0+3.  16 gate-rows of W_hh (bf16, packed
// pairs) live in LDS.  One device-wide barrier per timestep; h ping-pongs
// between hbuf[0]/hbuf[1]; h is staged into LDS via async-to-LDS copies.
__global__ __launch_bounds__(256)
void lstm_recurrent_kernel(const float* __restrict__ pre,    // [T,4096]
                           const float* __restrict__ w_hh,   // [4096,1024]
                           float* __restrict__ hs,           // [T,1024]
                           float* __restrict__ hbuf,         // [2,1024] (hbuf[0] pre-zeroed)
                           unsigned* __restrict__ bar,
                           int T) {
  __shared__ unsigned wlds[16 * H_DIM / 2];   // 32 KB: bf16 pairs, row-major
  __shared__ float    hlds[H_DIM];            // 4 KB broadcast h
  __shared__ float    gsm[16];                // per-row dot results
  __shared__ float    carr[REC_UNITS];        // cell state (block-private)

  const int tid  = threadIdx.x;
  const int j0   = blockIdx.x * REC_UNITS;
  const int lane = tid & 31;
  const int wave = tid >> 5;                  // 8 waves

  // Stage this block's 16 gate-rows (4 gates x 4 units) into LDS as bf16
  // pairs.  Vectorized: each thread converts 4 floats -> 2 packed uints.
  for (int idx = tid * 4; idx < 16 * H_DIM; idx += 256 * 4) {
    const int row = idx >> 10;                // row = g*4 + u
    const int g = row >> 2, u = row & 3;
    const int k = idx & (H_DIM - 1);
    const float4 w = *(const float4*)(w_hh + (size_t)(g * H_DIM + j0 + u) * H_DIM + k);
    uint2 p;
    p.x = pack_bf16x2(w.x, w.y);
    p.y = pack_bf16x2(w.z, w.w);
    *(uint2*)(wlds + (idx >> 1)) = p;
  }
  if (tid < REC_UNITS) carr[tid] = 0.0f;
  __syncthreads();

  unsigned step = 0;
  for (int t = 0; t < T; ++t) {
    // broadcast previous h (parity buffer written last step / zero-init)
    {
      const float* hsrc = hbuf + (t & 1) * H_DIM + tid * 4;
#if HAVE_ASYNC_COPY
      __builtin_amdgcn_global_load_async_to_lds_b128(
          (AS_GLOBAL v4i_t*)hsrc,
          (AS_SHARED v4i_t*)(hlds + tid * 4),
          0, 0);
#if HAVE_WAIT_ASYNC
      __builtin_amdgcn_s_wait_asynccnt(0);
#else
      asm volatile("s_wait_asynccnt 0x0" ::: "memory");
#endif
#else
      const float4 v = *(const float4*)hsrc;
      *(float4*)(hlds + tid * 4) = v;
#endif
    }
    __syncthreads();

    // dot products: 8 waves x 2 rows; each lane owns contiguous 8-elem
    // chunks (ds_load_b128 of packed bf16 + 2x ds_load_b128 of h).
#pragma unroll
    for (int rr = 0; rr < 2; ++rr) {
      const int row = wave * 2 + rr;
      const unsigned* wr = wlds + row * (H_DIM / 2);
      float acc = 0.0f;
#pragma unroll
      for (int pass = 0; pass < 4; ++pass) {
        const int kb = pass * 256 + lane * 8;       // 8 elements per lane
        const uint4  w  = *(const uint4*)(wr + (kb >> 1));
        const float4 h0 = *(const float4*)(hlds + kb);
        const float4 h1 = *(const float4*)(hlds + kb + 4);
        acc = fmaf(bf16_lo(w.x), h0.x, acc);
        acc = fmaf(bf16_hi(w.x), h0.y, acc);
        acc = fmaf(bf16_lo(w.y), h0.z, acc);
        acc = fmaf(bf16_hi(w.y), h0.w, acc);
        acc = fmaf(bf16_lo(w.z), h1.x, acc);
        acc = fmaf(bf16_hi(w.z), h1.y, acc);
        acc = fmaf(bf16_lo(w.w), h1.z, acc);
        acc = fmaf(bf16_hi(w.w), h1.w, acc);
      }
#pragma unroll
      for (int off = 16; off > 0; off >>= 1)
        acc += __shfl_down(acc, off, 32);
      if (lane == 0) gsm[row] = acc;
    }
    __syncthreads();

    // elementwise cell update for this block's 4 units
    if (tid < REC_UNITS) {
      const int u = tid;
      const float* pt = pre + (size_t)t * G_DIM + j0 + u;
      const float gi = pt[0 * H_DIM] + gsm[0 * REC_UNITS + u];
      const float gf = pt[1 * H_DIM] + gsm[1 * REC_UNITS + u];
      const float gg = pt[2 * H_DIM] + gsm[2 * REC_UNITS + u];
      const float go = pt[3 * H_DIM] + gsm[3 * REC_UNITS + u];
      const float i_ = sigmoidf_(gi);
      const float f_ = sigmoidf_(gf);
      const float g_ = tanhf(gg);
      const float o_ = sigmoidf_(go);
      const float c  = f_ * carr[u] + i_ * g_;
      carr[u] = c;
      const float h = o_ * tanhf(c);
      hbuf[((t + 1) & 1) * H_DIM + j0 + u] = h;
      hs[(size_t)t * H_DIM + j0 + u] = h;
    }

    // make h visible, then grid barrier (monotonic counter, zeroed per launch)
    __threadfence();
    ++step;
    if (tid == 0) {
      atomicAdd(bar, 1u);
      const unsigned target = step * (unsigned)REC_BLOCKS;
      while (__hip_atomic_load(bar, __ATOMIC_RELAXED, __HIP_MEMORY_SCOPE_AGENT) < target)
        __builtin_amdgcn_s_sleep(1);
    }
    __syncthreads();
    __threadfence();
  }
}

// ---------------- final linear: out[0] = hs2[T-1] . w_lin + b_lin ----------
__global__ __launch_bounds__(256)
void final_linear_kernel(const float* __restrict__ hs_last,
                         const float* __restrict__ w_lin,
                         const float* __restrict__ b_lin,
                         float* __restrict__ out) {
  __shared__ float red[256];
  const int tid = threadIdx.x;
  float acc = 0.0f;
  for (int k = tid; k < H_DIM; k += 256)
    acc = fmaf(hs_last[k], w_lin[k], acc);
  red[tid] = acc;
  __syncthreads();
  for (int s = 128; s > 0; s >>= 1) {
    if (tid < s) red[tid] += red[tid + s];
    __syncthreads();
  }
  if (tid == 0) out[0] = red[0] + b_lin[0];
}

// ---------------------------------------------------------------------------
extern "C" void kernel_launch(void* const* d_in, const int* in_sizes, int n_in,
                              void* d_out, int out_size, void* d_ws, size_t ws_size,
                              hipStream_t stream) {
  const float* seq = (const float*)d_in[0];
  const float* w_ih[3] = { (const float*)d_in[1], (const float*)d_in[5], (const float*)d_in[9]  };
  const float* w_hh[3] = { (const float*)d_in[2], (const float*)d_in[6], (const float*)d_in[10] };
  const float* b_ih[3] = { (const float*)d_in[3], (const float*)d_in[7], (const float*)d_in[11] };
  const float* b_hh[3] = { (const float*)d_in[4], (const float*)d_in[8], (const float*)d_in[12] };
  const float* w_lin = (const float*)d_in[13];
  const float* b_lin = (const float*)d_in[14];
  float* out = (float*)d_out;

  // ---- workspace layout (all 256B aligned) ----
  char* ws = (char*)d_ws;
  size_t off = 0;
  auto carve = [&](size_t bytes) { char* p = ws + off; off = (off + bytes + 255) & ~(size_t)255; return p; };
  unsigned* bars   = (unsigned*)carve(3 * 64);                              // 3 barrier slots
  float*    hbufs  = (float*)   carve(3 * 2 * H_DIM * sizeof(float));       // 3 layers x ping-pong h
  float*    pre    = (float*)   carve((size_t)T_STEPS * G_DIM * sizeof(float)); // 64 MB
  float*    hs0    = (float*)   carve((size_t)T_STEPS * H_DIM * sizeof(float)); // 16 MB
  float*    hs1    = (float*)   carve((size_t)T_STEPS * H_DIM * sizeof(float)); // 16 MB
  ushort_t* Abf    = (ushort_t*)carve((size_t)T_STEPS * H_DIM * sizeof(ushort_t)); // 8 MB
  ushort_t* Wbf    = (ushort_t*)carve((size_t)G_DIM  * H_DIM * sizeof(ushort_t)); // 8 MB
  (void)ws_size; (void)n_in; (void)in_sizes; (void)out_size;

  float* hs_buf[3] = { hs0, hs1, hs0 };   // layer2 reuses hs0

  // ---- init barriers + h buffers ----
  init_ws_kernel<<<(3 * 2 * H_DIM + 255) / 256, 256, 0, stream>>>(bars, hbufs);

  const float* layer_in = seq;
  for (int l = 0; l < 3; ++l) {
    const int K = (l == 0) ? IN_DIM : H_DIM;

    // cast activations and input weights to bf16
    {
      const int nA = T_STEPS * K;
      cast_f32_bf16_kernel<<<(nA + 255) / 256, 256, 0, stream>>>(layer_in, Abf, nA);
      const int nW = G_DIM * K;
      cast_f32_bf16_kernel<<<(nW + 255) / 256, 256, 0, stream>>>(w_ih[l], Wbf, nW);
    }

    // pre = Abf @ Wbf^T + (b_ih + b_hh)    [4096 x 4096], WMMA bf16
    {
      const int waves  = (T_STEPS / 16) * (G_DIM / 64);   // 16384 strips
      const int blocks = waves / 8;                        // 8 waves / block
      gemm_bf16_wmma_kernel<<<blocks, 256, 0, stream>>>(
          Abf, Wbf, b_ih[l], b_hh[l], pre, T_STEPS, G_DIM, K);
    }

    // sequential recurrence (persistent kernel, 1 grid barrier / step)
    lstm_recurrent_kernel<<<REC_BLOCKS, 256, 0, stream>>>(
        pre, w_hh[l], hs_buf[l], hbufs + l * 2 * H_DIM, bars + l * 16, T_STEPS);

    layer_in = hs_buf[l];
  }

  // out = hs2[T-1] . w_lin + b_lin
  final_linear_kernel<<<1, 256, 0, stream>>>(
      hs_buf[2] + (size_t)(T_STEPS - 1) * H_DIM, w_lin, b_lin, out);
}
